// GCN_31774168056026
// MI455X (gfx1250) — compile-verified, matrix operand order checked by reading, not run on hardware
//
#include <hip/hip_runtime.h>
#include <hip/hip_bf16.h>

#define D 96

typedef float    v2f  __attribute__((ext_vector_type(2)));
typedef float    v8f  __attribute__((ext_vector_type(8)));
typedef _Float16 v16h __attribute__((ext_vector_type(16)));

#ifndef __has_builtin
#define __has_builtin(x) 0
#endif

#if defined(__AMDGCN__) && __has_builtin(__builtin_amdgcn_wmma_f32_16x16x4_f32)
#define USE_F32_WMMA 1
#else
#define USE_F32_WMMA 0
#endif

// ---------------------------------------------------------------- utilities
__global__ void zero_kernel(float* __restrict__ p, size_t n) {
  size_t i = (size_t)blockIdx.x * blockDim.x + threadIdx.x;
  if (i < n) p[i] = 0.0f;
}

// deg[c] += 1 for every edge target; self-loop handled later as +1
__global__ void deg_kernel(const int* __restrict__ col, float* __restrict__ deg, int E) {
  int e = blockIdx.x * blockDim.x + threadIdx.x;
  if (e < E) atomicAdd(&deg[col[e]], 1.0f);
}

// dinv[i] = rsqrt(deg[i] + 1)   (always > 0 because of the self-loop)
__global__ void dinv_kernel(float* __restrict__ deg, int n) {
  int i = blockIdx.x * blockDim.x + threadIdx.x;
  if (i < n) deg[i] = rsqrtf(deg[i] + 1.0f);
}

// ---------------------------------------------------------------- WMMA GEMM
// C[M x 96] = A[M x 96] * B[96 x 96], one wave per 16x16 output tile.
// blockDim = 192 (6 waves) -> one block covers a full 16-row strip (6 N-tiles).
__global__ void gemm_wmma(const float* __restrict__ A, const float* __restrict__ B,
                          float* __restrict__ C, int M) {
  int m0 = blockIdx.x * 16;
  if (m0 + 16 > M) return;                 // wave-uniform; EXEC stays all-ones
  int wave = threadIdx.x >> 5;
  int lane = threadIdx.x & 31;
  int n0   = wave * 16;
  int l    = lane & 15;
  int half = lane >> 4;                    // 0: lanes 0-15, 1: lanes 16-31

  const float* arow = A + (size_t)(m0 + l) * D;
  v8f acc = {};

#if USE_F32_WMMA
  int kk = half * 2;                       // lanes 16-31 hold K=2,3 of each step
  #pragma unroll
  for (int k = 0; k < D; k += 4) {
    v2f a, b;
    a.x = arow[k + kk];
    a.y = arow[k + kk + 1];
    b.x = B[(size_t)(k + kk)     * D + n0 + l];
    b.y = B[(size_t)(k + kk + 1) * D + n0 + l];
    acc = __builtin_amdgcn_wmma_f32_16x16x4_f32(
        false, a, false, b, (short)0, acc, false, false);
  }
#else
  // fallback: fp16 inputs, fp32 accumulate (codegen-confirmed builtin)
  for (int k0 = 0; k0 < D; k0 += 32) {
    v16h a, b;
    #pragma unroll
    for (int e = 0; e < 16; ++e) {
      int kk = k0 + half * 8 + ((e >> 3) * 16) + (e & 7);   // §7.12.2 16-bit A layout
      a[e] = (_Float16)arow[kk];
      b[e] = (_Float16)B[(size_t)kk * D + n0 + l];
    }
    acc = __builtin_amdgcn_wmma_f32_16x16x32_f16(
        false, a, false, b, (short)0, acc, false, false);
  }
#endif

  // D-matrix layout: VGPR j -> row m0 + half*8 + j, col n0 + l
  int colIdx = n0 + l;
  int rbase  = m0 + half * 8;
  #pragma unroll
  for (int j = 0; j < 8; ++j)
    C[(size_t)(rbase + j) * D + colIdx] = acc[j];
}

// ------------------------------------------------------- edge scatter (atomics)
// one thread = one edge x 4 features; 24 threads per edge cover D=96
__global__ void scatter_edges(const int* __restrict__ row, const int* __restrict__ col,
                              const float* __restrict__ dinv, const float* __restrict__ h,
                              float* __restrict__ out, int E) {
  int idx = blockIdx.x * blockDim.x + threadIdx.x;
  int e = idx / 24;
  if (e >= E) return;
  int f = (idx % 24) * 4;
  int r = row[e], c = col[e];
  float nrm = dinv[r] * dinv[c];
  float4 hv = *(const float4*)(h + (size_t)r * D + f);
  float* o = out + (size_t)c * D + f;
  atomicAdd(o + 0, hv.x * nrm);
  atomicAdd(o + 1, hv.y * nrm);
  atomicAdd(o + 2, hv.z * nrm);
  atomicAdd(o + 3, hv.w * nrm);
}

// ------------------------------------------------- finalize: self-loop + bias (+relu)
// accOut[i][f] = (accOut[i][f] + h[i][f]*dinv[i]^2 + b[f]) , optionally relu'd
__global__ void finalize_kernel(float* __restrict__ accOut, const float* __restrict__ h,
                                const float* __restrict__ dinv, const float* __restrict__ b,
                                int relu, int N) {
  int idx = blockIdx.x * blockDim.x + threadIdx.x;
  if (idx >= N * 24) return;
  int node = idx / 24;
  int f = (idx % 24) * 4;
  float di = dinv[node];
  float s = di * di;
  size_t off = (size_t)node * D + f;
  float4 v  = *(const float4*)(accOut + off);
  float4 hv = *(const float4*)(h + off);
  float4 bv = *(const float4*)(b + f);
  v.x += hv.x * s + bv.x;
  v.y += hv.y * s + bv.y;
  v.z += hv.z * s + bv.z;
  v.w += hv.w * s + bv.w;
  if (relu) {
    v.x = fmaxf(v.x, 0.0f); v.y = fmaxf(v.y, 0.0f);
    v.z = fmaxf(v.z, 0.0f); v.w = fmaxf(v.w, 0.0f);
  }
  *(float4*)(accOut + off) = v;
}

// ---------------------------------------------------------------- launcher
extern "C" void kernel_launch(void* const* d_in, const int* in_sizes, int n_in,
                              void* d_out, int out_size, void* d_ws, size_t ws_size,
                              hipStream_t stream) {
  const float* x  = (const float*)d_in[0];
  const int*   ei = (const int*)d_in[1];
  const float* W1 = (const float*)d_in[2];
  const float* b1 = (const float*)d_in[3];
  const float* W2 = (const float*)d_in[4];
  const float* b2 = (const float*)d_in[5];
  float* out = (float*)d_out;

  int N = in_sizes[0] / D;   // 50000
  int E = in_sizes[1] / 2;   // 800000
  const int* row = ei;
  const int* col = ei + E;

  float* ws   = (float*)d_ws;
  float* dinv = ws;                                 // N floats
  float* h    = ws + (((size_t)N + 127) & ~127ull); // N*D floats
  float* acc  = h + (size_t)N * D;                  // N*D floats

  const int tpb = 256;
  size_t nd = (size_t)N * D;
  int gN    = (N + tpb - 1) / tpb;
  int gE    = (E + tpb - 1) / tpb;
  int gND   = (int)((nd + tpb - 1) / tpb);
  int gEf   = (int)(((size_t)E * 24 + tpb - 1) / tpb);
  int gNf   = (N * 24 + tpb - 1) / tpb;
  int gGemm = N / 16;                               // 50000 is a multiple of 16

  // normalization
  zero_kernel<<<gN, tpb, 0, stream>>>(dinv, (size_t)N);
  deg_kernel<<<gE, tpb, 0, stream>>>(col, dinv, E);
  dinv_kernel<<<gN, tpb, 0, stream>>>(dinv, N);

  // layer 1: h = x@W1 ; acc = scatter(norm*h) ; acc = relu(acc + h*dinv^2 + b1)
  zero_kernel<<<gND, tpb, 0, stream>>>(acc, nd);
  gemm_wmma<<<gGemm, 192, 0, stream>>>(x, W1, h, N);
  scatter_edges<<<gEf, tpb, 0, stream>>>(row, col, dinv, h, acc, E);
  finalize_kernel<<<gNf, tpb, 0, stream>>>(acc, h, dinv, b1, 1, N);

  // layer 2: h = acc@W2 ; out = scatter(norm*h) ; out = out + h*dinv^2 + b2
  gemm_wmma<<<gGemm, 192, 0, stream>>>(acc, W2, h, N);
  zero_kernel<<<gND, tpb, 0, stream>>>(out, nd);
  scatter_edges<<<gEf, tpb, 0, stream>>>(row, col, dinv, h, out, E);
  finalize_kernel<<<gNf, tpb, 0, stream>>>(out, h, dinv, b2, 0, N);
}
